// TimeRecurrentCell_712964571376
// MI455X (gfx1250) — compile-verified, matrix operand order checked by reading, not run on hardware
//
#include <hip/hip_runtime.h>

// LSTM scan, persistent-kernel design for MI455X (gfx1250):
//   gates[t] = [x_t | h_{t-1}] (32x2048) @ Wcat^T (2048x4096) + bias
// 64 workgroups, each owns 16 hidden units (=64 gate columns across i,f,g,o).
// Per-WG weight slice (64 rows x 2048 K, bf16 = 256 KB) lives in LDS for the
// whole scan. WMMA bf16 16x16x32, f32 accumulate. One L2-resident grid
// barrier per timestep; h double-buffered in bf16 (128 KB, L2-resident).

#define B_  32
#define T_  2048
#define D_  1024
#define H_  1024
#define G4  (4 * H_)          // 4096 gate columns
#define KTOT (D_ + H_)        // 2048 fused K
#define NWG 64
#define CPW 16                // hidden units per WG
#define THREADS 256

typedef __attribute__((ext_vector_type(16))) __bf16 v16bf;
typedef __attribute__((ext_vector_type(8)))  __bf16 v8bf;
typedef __attribute__((ext_vector_type(8)))  float  v8f;

// ---- workspace layout (bytes) ----
#define WCAT_BYTES ((size_t)G4 * KTOT * 2)            // 16 MB bf16 [4096][2048]
#define BIAS_OFF   WCAT_BYTES
#define BIAS_BYTES ((size_t)G4 * 4)                   // 16 KB f32
#define HBUF_OFF   (BIAS_OFF + BIAS_BYTES)
#define HBUF_BYTES ((size_t)2 * B_ * H_ * 2)          // 128 KB bf16 [2][32][1024]
#define BAR_OFF    (HBUF_OFF + HBUF_BYTES)

// ---------------- prep: Wcat bf16, bias sum, zero h, reset barrier ----------
__global__ void lstm_prep(const float* __restrict__ Wih,
                          const float* __restrict__ Whh,
                          const float* __restrict__ bih,
                          const float* __restrict__ bhh,
                          __bf16* __restrict__ Wcat,
                          float*  __restrict__ bias,
                          __bf16* __restrict__ hbuf,
                          unsigned* __restrict__ bar) {
  long idx = (long)blockIdx.x * THREADS + threadIdx.x;
  long total = (long)G4 * KTOT;
  if (idx < total) {
    int n = (int)(idx / KTOT), k = (int)(idx % KTOT);
    float w = (k < D_) ? Wih[(long)n * D_ + k] : Whh[(long)n * H_ + (k - D_)];
    Wcat[idx] = (__bf16)w;
  }
  if (idx < G4)          bias[idx] = bih[idx] + bhh[idx];
  if (idx < 2 * B_ * H_) hbuf[idx] = (__bf16)0.0f;
  if (idx < 2)           bar[idx] = 0u;
}

__device__ inline v16bf cat8(v8bf lo, v8bf hi) {
  return __builtin_shufflevector(lo, hi, 0,1,2,3,4,5,6,7,8,9,10,11,12,13,14,15);
}

// ---------------- persistent scan kernel ----------------
__launch_bounds__(THREADS, 1)
__global__ void lstm_scan(const float* __restrict__ X,
                          const __bf16* __restrict__ Wcat,
                          const float*  __restrict__ bias,
                          __bf16* __restrict__ hbuf,
                          float*  __restrict__ out,
                          unsigned* __restrict__ bar) {
  extern __shared__ __bf16 Wlds[];                 // [64][KTOT] = 256 KB
  __shared__ float gates[B_][4][CPW];              // 8 KB
  __shared__ float cst[B_][CPW];                   // 2 KB cell state
  __shared__ float blds[4][CPW];                   // bias slice

  const int wg   = blockIdx.x;                     // 0..63
  const int tid  = threadIdx.x;
  const int lane = tid & 31;
  const int wv   = tid >> 5;                       // wave 0..7
  const int mtile = wv >> 2;                       // 0..1  (batch tile)
  const int nt    = wv & 3;                        // 0..3  (gate = N tile)
  const int half  = lane >> 4;                     // K-half select (ISA layout)
  const int l15   = lane & 15;

  // --- stage this WG's 64 weight rows (gate g, unit j) into LDS, 16B chunks
  for (int idx = tid; idx < 64 * (KTOT / 8); idx += THREADS) {
    int r   = idx >> 8;                            // local row 0..63
    int off = (idx & 255) * 8;                     // bf16 offset in row
    int g = r >> 4, j = r & 15;
    const v8bf* src =
        (const v8bf*)(Wcat + ((long)(g * H_ + wg * CPW + j)) * KTOT + off);
    *(v8bf*)(Wlds + (long)r * KTOT + off) = *src;
  }
  if (tid < 4 * CPW) {
    int g = tid >> 4, j = tid & 15;
    blds[g][j] = bias[g * H_ + wg * CPW + j];
  }
  for (int idx = tid; idx < B_ * CPW; idx += THREADS)
    cst[idx >> 4][idx & 15] = 0.0f;
  __syncthreads();

  const int m = mtile * 16 + l15;                  // batch row owned by lane
  const __bf16* wrow = Wlds + (long)(nt * 16 + l15) * KTOT;  // B column (gate col)

  for (int t = 0; t < T_; ++t) {
    v8f acc = {};
    const float* xrow = X + ((long)m * T_ + t) * D_;
    if (t + 1 < T_) __builtin_prefetch(xrow + D_, 0, 1);  // next step's x row

    // ---- phase 1: K = [0,1024)  A from X (fp32 -> bf16 on the fly)
#pragma unroll 4
    for (int kk = 0; kk < D_ / 32; ++kk) {
      int k0 = kk * 32 + half * 8;
      const float4* p0 = (const float4*)(xrow + k0);
      const float4* p1 = (const float4*)(xrow + k0 + 16);
      float4 a0 = p0[0], a1 = p0[1], b0 = p1[0], b1 = p1[1];
      v16bf afrag = { (__bf16)a0.x, (__bf16)a0.y, (__bf16)a0.z, (__bf16)a0.w,
                      (__bf16)a1.x, (__bf16)a1.y, (__bf16)a1.z, (__bf16)a1.w,
                      (__bf16)b0.x, (__bf16)b0.y, (__bf16)b0.z, (__bf16)b0.w,
                      (__bf16)b1.x, (__bf16)b1.y, (__bf16)b1.z, (__bf16)b1.w };
      v8bf w0 = *(const v8bf*)(wrow + kk * 32 + half * 8);
      v8bf w1 = *(const v8bf*)(wrow + kk * 32 + 16 + half * 8);
      acc = __builtin_amdgcn_wmma_f32_16x16x32_bf16(
          false, afrag, false, cat8(w0, w1), (short)0, acc, false, false);
    }

    // ---- phase 2: K = [1024,2048)  A from bf16 h double buffer
    const __bf16* hrow = hbuf + ((long)(t & 1) * B_ + m) * H_;
    const __bf16* wrow2 = wrow + D_;
#pragma unroll 4
    for (int kk = 0; kk < H_ / 32; ++kk) {
      int k0 = kk * 32 + half * 8;
      v8bf h0 = *(const v8bf*)(hrow + k0);
      v8bf h1 = *(const v8bf*)(hrow + k0 + 16);
      v8bf w0 = *(const v8bf*)(wrow2 + kk * 32 + half * 8);
      v8bf w1 = *(const v8bf*)(wrow2 + kk * 32 + 16 + half * 8);
      acc = __builtin_amdgcn_wmma_f32_16x16x32_bf16(
          false, cat8(h0, h1), false, cat8(w0, w1), (short)0, acc, false, false);
    }

    // ---- scatter D tile to LDS (ISA C/D layout: M = r + 8*half, N = l15)
#pragma unroll
    for (int r = 0; r < 8; ++r)
      gates[mtile * 16 + half * 8 + r][nt][l15] = acc[r];
    __syncthreads();

    // ---- elementwise LSTM cell update for this WG's 16 hidden units
    for (int idx = tid; idx < B_ * CPW; idx += THREADS) {
      int mb = idx >> 4, j = idx & 15;
      float ig = gates[mb][0][j] + blds[0][j];
      float fg = gates[mb][1][j] + blds[1][j];
      float gg = gates[mb][2][j] + blds[2][j];
      float og = gates[mb][3][j] + blds[3][j];
      float i = 1.0f / (1.0f + __expf(-ig));
      float f = 1.0f / (1.0f + __expf(-fg));
      float g = tanhf(gg);
      float o = 1.0f / (1.0f + __expf(-og));
      float c = f * cst[mb][j] + i * g;
      cst[mb][j] = c;
      float h = o * tanhf(c);
      int col = wg * CPW + j;
      out[((long)mb * T_ + t) * H_ + col] = h;                       // outputs
      hbuf[((long)((t + 1) & 1) * B_ + mb) * H_ + col] = (__bf16)h;  // next A
      if (t == T_ - 1) {
        out[(long)B_ * T_ * H_ + (long)mb * H_ + col] = h;           // h_T
        out[(long)B_ * T_ * H_ + (long)B_ * H_ + (long)mb * H_ + col] = c; // c_T
      }
    }

    // ---- grid barrier (sense/generation counter, L2-resident)
    __threadfence();
    __syncthreads();
    if (tid == 0) {
      unsigned gen = atomicAdd(&bar[1], 0u);
      if (atomicAdd(&bar[0], 1u) == NWG - 1u) {
        atomicExch(&bar[0], 0u);
        __threadfence();
        atomicAdd(&bar[1], 1u);
      } else {
        while (atomicAdd(&bar[1], 0u) == gen) __builtin_amdgcn_s_sleep(2);
      }
    }
    __syncthreads();
  }
}

extern "C" void kernel_launch(void* const* d_in, const int* in_sizes, int n_in,
                              void* d_out, int out_size, void* d_ws, size_t ws_size,
                              hipStream_t stream) {
  (void)in_sizes; (void)n_in; (void)out_size; (void)ws_size;
  const float* X   = (const float*)d_in[0];
  const float* Wih = (const float*)d_in[1];
  const float* Whh = (const float*)d_in[2];
  const float* bih = (const float*)d_in[3];
  const float* bhh = (const float*)d_in[4];

  char* ws = (char*)d_ws;
  __bf16*   Wcat = (__bf16*)(ws);
  float*    bias = (float*)(ws + BIAS_OFF);
  __bf16*   hbuf = (__bf16*)(ws + HBUF_OFF);
  unsigned* bar  = (unsigned*)(ws + BAR_OFF);

  long total = (long)G4 * KTOT;
  int pblocks = (int)((total + THREADS - 1) / THREADS);
  lstm_prep<<<pblocks, THREADS, 0, stream>>>(Wih, Whh, bih, bhh,
                                             Wcat, bias, hbuf, bar);

  size_t dyn_lds = (size_t)64 * KTOT * sizeof(__bf16);   // 256 KB weight slice
  lstm_scan<<<NWG, THREADS, dyn_lds, stream>>>(X, Wcat, bias, hbuf,
                                               (float*)d_out, bar);
}
